// MultiHeadMultiGran3Attention_78632261255592
// MI455X (gfx1250) — matrix operand order repaired and checked by reading, not compiled, not run on hardware
//
#include <hip/hip_runtime.h>

// ---------------- CDNA5 WMMA types ----------------
typedef __attribute__((ext_vector_type(16))) __bf16 v16bf;
typedef __attribute__((ext_vector_type(8)))  float  v8f;

union BF16Frag {
  v16bf v;
  unsigned short u[16];
  uint4 q[2];
};

__device__ __forceinline__ unsigned short f2bf(float f){
  unsigned u = __float_as_uint(f);
  u += 0x7fffu + ((u >> 16) & 1u);            // round-to-nearest-even
  return (unsigned short)(u >> 16);
}

__device__ __forceinline__ v8f vzero8(){
  v8f z;
  #pragma unroll
  for (int i=0;i<8;i++) z[i]=0.0f;
  return z;
}

__device__ __forceinline__ v8f wmma_bf16(const BF16Frag& a, const BF16Frag& b, v8f c){
  // D = A(16x32 bf16) * B(32x16 bf16) + C(16x16 f32)
  return __builtin_amdgcn_wmma_f32_16x16x32_bf16(false, a.v, false, b.v, (short)0, c, false, false);
}

// -------- gfx1250 async global->LDS copy (ASYNCcnt path), with fallback ----
#if defined(__gfx1250__) && __has_builtin(__builtin_amdgcn_global_load_async_to_lds_b128)
#define HAVE_ASYNC_LDS 1
#else
#define HAVE_ASYNC_LDS 0
#endif

// builtin parameter type (from hipcc diagnostic): int __vector(4) __device__* / AS3*
typedef int async_vec __attribute__((vector_size(16)));

__device__ __forceinline__ void copy_g2l_b128(const unsigned short* g, unsigned short* l){
#if HAVE_ASYNC_LDS
  async_vec* gp = (async_vec*)g;   // strip const, retype
  async_vec* lp = (async_vec*)l;
  __builtin_amdgcn_global_load_async_to_lds_b128(
      (__attribute__((address_space(1))) async_vec*)gp,
      (__attribute__((address_space(3))) async_vec*)lp, 0, 0);
#else
  *(uint4*)l = *(const uint4*)g;
#endif
}

__device__ __forceinline__ void wait_async_copies(){
#if HAVE_ASYNC_LDS
#if __has_builtin(__builtin_amdgcn_s_wait_asynccnt)
  __builtin_amdgcn_s_wait_asynccnt(0);
#else
  asm volatile("s_wait_asynccnt 0x0" ::: "memory");
#endif
#endif
}

// ---------------- problem constants ----------------
#define DIMC   1024
#define NH     16
#define HDIM   64
#define BATCH  8
#define SLEN   512
#define TLEN   256
#define MLEN   128
#define STOT   896
#define QT_N   56     // 896/16
#define KT32_N 28     // 896/32
#define NROWS  (BATCH*STOT)   // 7168

// fragment-layout index helpers (CDNA5 ISA 7.12.2, wave32)
// A (16x32, 16-bit): lane row m = lane&15, group g = lane>>4,
//   element e -> K = (e<8) ? 8g+e : 16+8g+(e-8)
// B (32x16, 16-bit): lane col n = lane&15, group g, element e -> K as A; attn packs A,B with K = e + 16g
// C (16x16, f32): VGPR v -> row = v + 8g, col = lane&15

// ===================================================================
// Pack kernel A: x = concat(stok,tok,stm) -> bf16 row-major [7168][1024]
// ===================================================================
__global__ __launch_bounds__(256) void pack_x_kernel(
    const float* __restrict__ stok, const float* __restrict__ tok,
    const float* __restrict__ stm, unsigned short* __restrict__ xbf)
{
  size_t t = (size_t)blockIdx.x*256 + threadIdx.x;   // 917504 threads
  size_t base = t*8;
  int row = (int)(base >> 10);
  int col = (int)(base & 1023);
  int b = row / STOT, s = row % STOT;
  const float* src;
  if (s < SLEN)           src = stok + ((size_t)b*SLEN + s)*DIMC + col;
  else if (s < SLEN+TLEN) src = tok  + ((size_t)b*TLEN + (s-SLEN))*DIMC + col;
  else                    src = stm  + ((size_t)b*MLEN + (s-SLEN-TLEN))*DIMC + col;
  unsigned short tmp[8];
  #pragma unroll
  for (int e=0;e<8;e++) tmp[e] = f2bf(src[e]);
  *(uint4*)(xbf + base) = *(const uint4*)tmp;
}

// ===================================================================
// Pack kernel B: transpose weights to bf16  W^T[n][k]
//   WqT[3072][1024] from Wqkv[1024][3072]; WoT[1024][1024] from Wout
// ===================================================================
__global__ __launch_bounds__(256) void pack_w_kernel(
    const float* __restrict__ Wqkv, const float* __restrict__ Wout,
    unsigned short* __restrict__ WqT, unsigned short* __restrict__ WoT)
{
  size_t t = (size_t)blockIdx.x*256 + threadIdx.x;    // 524288 threads
  const size_t NQ = (size_t)3*DIMC*DIMC/8;            // 393216
  unsigned short tmp[8];
  if (t < NQ){
    size_t base = t*8;
    int n = (int)(base >> 10), k0 = (int)(base & 1023);
    #pragma unroll
    for (int e=0;e<8;e++) tmp[e] = f2bf(Wqkv[(size_t)(k0+e)*(3*DIMC) + n]);
    *(uint4*)(WqT + base) = *(const uint4*)tmp;
  } else {
    size_t base = (t - NQ)*8;
    int n = (int)(base >> 10), k0 = (int)(base & 1023);
    #pragma unroll
    for (int e=0;e<8;e++) tmp[e] = f2bf(Wout[(size_t)(k0+e)*DIMC + n]);
    *(uint4*)(WoT + base) = *(const uint4*)tmp;
  }
}

// ===================================================================
// Pack kernel C: relative-position embedding tables into B-fragments
//   EmbK: prel = q @ emb_k^T  ->  B[k=hd (2x32)][n=bin (5x16)]
//   EmbV: ctx += w @ emb_v    ->  B[k=bin (3x32)][n=hd (4x16)]
// ===================================================================
__global__ __launch_bounds__(256) void pack_emb_kernel(
    const float* __restrict__ stok_k, const float* __restrict__ tok_k, const float* __restrict__ stm_k,
    const float* __restrict__ stok_v, const float* __restrict__ tok_v, const float* __restrict__ stm_v,
    unsigned short* __restrict__ EmbK, unsigned short* __restrict__ EmbV)
{
  const int NK = 3*2*5*512;
  const int NV = 3*3*4*512;
  int idx = blockIdx.x*256 + threadIdx.x;
  if (idx < NK){
    int e    = idx & 15;
    int lane = (idx >> 4) & 31;
    int frag = idx >> 9;
    int t    = frag % 5;
    int c    = (frag / 5) % 2;
    int seg  = frag / 10;
    int nn = lane & 15, gg = lane >> 4;
    int bin = t*16 + nn;
    int hd  = c*32 + 16*gg + e;
    int vocab = (seg == 2) ? 33 : 65;
    const float* tab = (seg==0) ? stok_k : ((seg==1) ? tok_k : stm_k);
    float val = (bin < vocab) ? tab[bin*HDIM + hd] : 0.0f;
    EmbK[idx] = f2bf(val);
  } else if (idx < NK + NV){
    int j    = idx - NK;
    int e    = j & 15;
    int lane = (j >> 4) & 31;
    int frag = j >> 9;
    int nt   = frag % 4;
    int kc   = (frag / 4) % 3;
    int seg  = frag / 12;
    int nn = lane & 15, gg = lane >> 4;
    int bin = kc*32 + 16*gg + e;
    int hd  = nt*16 + nn;
    int vocab = (seg == 2) ? 33 : 65;
    const float* tab = (seg==0) ? stok_v : ((seg==1) ? tok_v : stm_v);
    float val = (bin < vocab) ? tab[bin*HDIM + hd] : 0.0f;
    EmbV[j] = f2bf(val);
  }
}

// ===================================================================
// Shared bf16 WMMA GEMM mainloop, double-buffered LDS staging via
// async global->LDS copies (no VGPR data movement in the hot loop).
// Block = 128 threads = 4 waves (2x2), block tile 64x128, wave tile 32x64.
// A: bf16 row-major [M][K], lda = K = 1024.  B: bf16 TRANSPOSED [N][K].
// LDS (ushort units): Asbuf(b)=b*2560 (64x40), Bsbuf(b)=5120+b*5120 (128x40)
// ===================================================================
#define SMEM_STAGE_USH 15360   // 2*(64*40) + 2*(128*40)

__device__ __forceinline__ void gemm_mainloop_bf(
    const unsigned short* __restrict__ Ablk,   // + row*1024
    const unsigned short* __restrict__ Bblk,   // + col*1024 (transposed)
    unsigned short* smem, v8f acc[2][4])
{
  const int tid  = threadIdx.x;
  const int wv   = tid >> 5;
  const int lane = tid & 31;
  const int wm = (wv & 1) * 32;
  const int wn = (wv >> 1) * 64;
  const int ln = lane & 15;
  const int g  = lane >> 4;

  // per-thread staging coords
  const int ar = tid >> 2, aq = tid & 3;            // A: rows 0..31 (+32), quad
  const int bc = tid >> 2, bq = tid & 3;            // B: cols 0..31 (+32,64,96), quad

  // ---- stage chunk 0 into buffer 0 ----
  #pragma unroll
  for (int i=0;i<2;i++)
    copy_g2l_b128(Ablk + (size_t)(ar + i*32)*DIMC + aq*8,
                  smem + (ar + i*32)*40 + aq*8);
  #pragma unroll
  for (int i=0;i<4;i++)
    copy_g2l_b128(Bblk + (size_t)(bc + i*32)*DIMC + bq*8,
                  smem + 5120 + (bc + i*32)*40 + bq*8);

  int p = 0;
  for (int kc = 0; kc < DIMC; kc += 32){
    const bool more = (kc + 32) < DIMC;
    wait_async_copies();     // our async writes into buf p complete
    __syncthreads();         // ... and everyone else's
    // issue next chunk's async copies into the other buffer (overlaps WMMAs)
    if (more){
      unsigned short* Asn = smem + (p^1)*2560;
      unsigned short* Bsn = smem + 5120 + (p^1)*5120;
      #pragma unroll
      for (int i=0;i<2;i++)
        copy_g2l_b128(Ablk + (size_t)(ar + i*32)*DIMC + kc + 32 + aq*8,
                      Asn + (ar + i*32)*40 + aq*8);
      #pragma unroll
      for (int i=0;i<4;i++)
        copy_g2l_b128(Bblk + (size_t)(bc + i*32)*DIMC + kc + 32 + bq*8,
                      Bsn + (bc + i*32)*40 + bq*8);
    }
    // fragment loads from buffer p
    const unsigned short* Asb = smem + p*2560;
    const unsigned short* Bsb = smem + 5120 + p*5120;
    BF16Frag a[2], b[4];
    #pragma unroll
    for (int mf=0; mf<2; mf++){
      const unsigned short* q0 = Asb + (wm + mf*16 + ln)*40;
      a[mf].q[0] = *(const uint4*)(q0 + 8*g);
      a[mf].q[1] = *(const uint4*)(q0 + 16 + 8*g);
    }
    #pragma unroll
    for (int nf=0; nf<4; nf++){
      const unsigned short* q0 = Bsb + (wn + nf*16 + ln)*40;
      b[nf].q[0] = *(const uint4*)(q0 + 8*g);
      b[nf].q[1] = *(const uint4*)(q0 + 16 + 8*g);
    }
    #pragma unroll
    for (int mf=0; mf<2; mf++)
      #pragma unroll
      for (int nf=0; nf<4; nf++)
        acc[mf][nf] = wmma_bf16(a[mf], b[nf], acc[mf][nf]);
    p ^= 1;
  }
}

// ===================================================================
// Kernel: QKV GEMM (bf16 operands). Writes Q as A-frags (pre-scaled 1/8),
// K,V as B-frags for the attention kernel.   grid = (112, 24), block = 128
// ===================================================================
__global__ __launch_bounds__(128) void gemm_qkv_kernel(
    const unsigned short* __restrict__ xbf, const unsigned short* __restrict__ WqT,
    const float* __restrict__ bqkv,
    unsigned short* __restrict__ Qp, unsigned short* __restrict__ Kp, unsigned short* __restrict__ Vp)
{
  __shared__ __align__(16) char smraw[4*32*66*4];   // max(staging 30720, Cs 33792)
  unsigned short* smem = (unsigned short*)smraw;
  float* Cs = (float*)smraw;                        // [4][32][66] after mainloop

  const int mb = blockIdx.x, nb = blockIdx.y;
  const unsigned short* Ablk = xbf + (size_t)mb*64*DIMC;
  const unsigned short* Bblk = WqT + (size_t)nb*128*DIMC;

  v8f acc[2][4];
  #pragma unroll
  for (int mf=0; mf<2; mf++)
    #pragma unroll
    for (int nf=0; nf<4; nf++) acc[mf][nf] = vzero8();

  gemm_mainloop_bf(Ablk, Bblk, smem, acc);

  const int tid=threadIdx.x, wv=tid>>5, lane=tid&31, ln=lane&15, g=lane>>4;
  const int wm=(wv&1)*32, wn=(wv>>1)*64;

  __syncthreads();   // staging done; reuse LDS as Cs
  #pragma unroll
  for (int mf=0; mf<2; mf++)
    #pragma unroll
    for (int nf=0; nf<4; nf++)
      #pragma unroll
      for (int v=0; v<8; v++)
        Cs[wv*2112 + (mf*16 + v + 8*g)*66 + nf*16 + ln] = acc[mf][nf][v];
  __syncthreads();

  const int col0 = nb*128 + wn;         // multiple of 64 -> one full head
  const int sel  = col0 / DIMC;         // 0=q, 1=k, 2=v
  const int h    = (col0 % DIMC) / HDIM;
  const int rw   = mb*64 + wm;          // multiple of 32
  const int bb   = rw / STOT;
  const int sw   = rw % STOT;
  const float* Cw = Cs + wv*2112;

  if (sel == 0){
    #pragma unroll
    for (int c=0; c<2; c++)
      #pragma unroll
      for (int sub=0; sub<2; sub++){
        int qt = (sw >> 4) + sub;
        unsigned short* dst = Qp + ((size_t)(((bb*NH + h)*QT_N + qt)*2 + c))*512 + lane*16;
        #pragma unroll
        for (int e=0;e<16;e++){
          int kk = e + 16*g;
          float val = (Cw[(sub*16 + ln)*66 + c*32 + kk] + bqkv[col0 + c*32 + kk]) * 0.125f;
          dst[e] = f2bf(val);
        }
      }
  } else if (sel == 1){
    #pragma unroll
    for (int c=0; c<2; c++)
      #pragma unroll
      for (int sub=0; sub<2; sub++){
        int kt = (sw >> 4) + sub;
        unsigned short* dst = Kp + ((size_t)(((bb*NH + h)*QT_N + kt)*2 + c))*512 + lane*16;
        #pragma unroll
        for (int e=0;e<16;e++){
          int kk = e + 16*g;
          float val = Cw[(sub*16 + ln)*66 + c*32 + kk] + bqkv[col0 + c*32 + kk];
          dst[e] = f2bf(val);
        }
      }
  } else {
    int kt32 = sw >> 5;
    #pragma unroll
    for (int nc=0; nc<4; nc++){
      unsigned short* dst = Vp + ((size_t)(((bb*NH + h)*KT32_N + kt32)*4 + nc))*512 + lane*16;
      #pragma unroll
      for (int e=0;e<16;e++){
        int kk = e + 16*g;
        float val = Cw[kk*66 + nc*16 + ln] + bqkv[col0 + nc*16 + ln];
        dst[e] = f2bf(val);
      }
    }
  }
}

// ===================================================================
// Kernel: attention. One wave per (b, h, 16-query tile); flash softmax
// over 28 key blocks of 32; rel-pos via prel lookup + LDS bin scatter +
// w @ emb_v WMMA.   grid = 1792, block = 128 (4 waves)
// ===================================================================
__global__ __launch_bounds__(128) void attn_kernel(
    const unsigned short* __restrict__ Qp, const unsigned short* __restrict__ Kp,
    const unsigned short* __restrict__ Vp,
    const unsigned short* __restrict__ EmbK, const unsigned short* __restrict__ EmbV,
    const unsigned char* __restrict__ mk0, const unsigned char* __restrict__ mk1,
    const unsigned char* __restrict__ mk2,
    unsigned short* __restrict__ ctxbuf)
{
  __shared__ float prel[4][16*80];
  __shared__ float wbin[4][16*96];
  __shared__ __align__(16) unsigned short pst[4][16*32];
  __shared__ float alph[4][16];

  const int tid  = threadIdx.x;
  const int wv   = tid >> 5;
  const int lane = tid & 31;
  const int wid  = blockIdx.x*4 + wv;
  const int b   = wid / (NH*QT_N);
  const int rem = wid % (NH*QT_N);
  const int h   = rem / QT_N;
  const int qt  = rem % QT_N;
  const int qbase = qt*16;
  const int ln = lane & 15, g = lane >> 4;

  int seg, s0, slen, R;
  if (qbase < SLEN)            { seg=0; s0=0;         slen=SLEN; R=32; }
  else if (qbase < SLEN+TLEN)  { seg=1; s0=SLEN;      slen=TLEN; R=32; }
  else                         { seg=2; s0=SLEN+TLEN; slen=MLEN; R=16; }

  for (int i=lane; i<16*96; i+=32) wbin[wv][i] = 0.0f;

  BF16Frag qf[2];
  {
    const uint4* qs = (const uint4*)(Qp + ((size_t)(((b*NH + h)*QT_N + qt)*2))*512);
    qf[0].q[0] = qs[lane*2];      qf[0].q[1] = qs[lane*2 + 1];
    qf[1].q[0] = qs[64 + lane*2]; qf[1].q[1] = qs[64 + lane*2 + 1];
  }

  // prel[i, bin] = q_i . emb_k[bin]  (5 n-tiles always, for block uniformity)
  #pragma unroll
  for (int t=0; t<5; t++){
    v8f acc = vzero8();
    #pragma unroll
    for (int c=0; c<2; c++){
      const uint4* es = (const uint4*)(EmbK + ((size_t)((seg*2 + c)*5 + t))*512);
      BF16Frag ek; ek.q[0] = es[lane*2]; ek.q[1] = es[lane*2+1];
      acc = wmma_bf16(qf[c], ek, acc);
    }
    #pragma unroll
    for (int v=0; v<8; v++)
      prel[wv][(v + 8*g)*80 + t*16 + ln] = acc[v];
  }

  float mrow[8], lrow[8];
  #pragma unroll
  for (int v=0; v<8; v++){ mrow[v] = -3.0e38f; lrow[v] = 0.0f; }
  v8f ctx[4];
  #pragma unroll
  for (int nc=0; nc<4; nc++) ctx[nc] = vzero8();

  __syncthreads();

  for (int kb=0; kb<KT32_N; kb++){
    const int kbase = kb*32;
    const bool inseg = (kbase >= s0) && (kbase < s0 + slen);

    v8f sfr[2];
    #pragma unroll
    for (int sub=0; sub<2; sub++){
      const uint4* ks = (const uint4*)(Kp + ((size_t)(((b*NH + h)*QT_N + (kb*2 + sub))*2))*512);
      BF16Frag k0, k1;
      k0.q[0] = ks[lane*2];      k0.q[1] = ks[lane*2 + 1];
      k1.q[0] = ks[64 + lane*2]; k1.q[1] = ks[64 + lane*2 + 1];
      v8f acc = vzero8();
      acc = wmma_bf16(qf[0], k0, acc);
      acc = wmma_bf16(qf[1], k1, acc);
      sfr[sub] = acc;
    }

    #pragma unroll
    for (int sub=0; sub<2; sub++){
      #pragma unroll
      for (int v=0; v<8; v++){
        int r = v + 8*g;
        int i = qbase + r;
        int j = kbase + sub*16 + ln;
        float s = sfr[sub][v];
        if (inseg){
          int d = j - i;
          d = (d < -R) ? -R : ((d > R) ? R : d);
          s += prel[wv][r*80 + d + R];
        }
        unsigned char mbyte;
        if (j < SLEN)           mbyte = mk0[b*SLEN + j];
        else if (j < SLEN+TLEN) mbyte = mk1[b*TLEN + (j - SLEN)];
        else                    mbyte = mk2[b*MLEN + (j - SLEN - TLEN)];
        if (mbyte) s = -1.0e18f;
        sfr[sub][v] = s;
      }
    }

    float alpha[8];
    #pragma unroll
    for (int v=0; v<8; v++){
      float x = fmaxf(sfr[0][v], sfr[1][v]);
      #pragma unroll
      for (int off=1; off<16; off<<=1)
        x = fmaxf(x, __shfl_xor(x, off, 32));
      float mnew = fmaxf(mrow[v], x);
      alpha[v] = __expf(mrow[v] - mnew);
      mrow[v] = mnew;
    }
    if (ln == 0){
      #pragma unroll
      for (int v=0; v<8; v++) alph[wv][8*g + v] = alpha[v];
    }
    __syncthreads();   // B1: alphas visible

    #pragma unroll
    for (int v=0; v<8; v++) lrow[v] *= alpha[v];
    #pragma unroll
    for (int nc=0; nc<4; nc++)
      #pragma unroll
      for (int v=0; v<8; v++) ctx[nc][v] *= alpha[v];
    for (int i=lane; i<16*96; i+=32){
      int r = i / 96;
      wbin[wv][i] *= alph[wv][r];
    }

    #pragma unroll
    for (int v=0; v<8; v++){
      int r = v + 8*g;
      float p0 = __expf(sfr[0][v] - mrow[v]);
      float p1 = __expf(sfr[1][v] - mrow[v]);
      sfr[0][v] = p0; sfr[1][v] = p1;
      pst[wv][r*32 + ln]      = f2bf(p0);
      pst[wv][r*32 + 16 + ln] = f2bf(p1);
      float t = p0 + p1;
      #pragma unroll
      for (int off=1; off<16; off<<=1)
        t += __shfl_xor(t, off, 32);
      lrow[v] += t;
    }
    __syncthreads();   // B2: pstage + wbin rescale complete

    if (inseg){
      #pragma unroll
      for (int sub=0; sub<2; sub++){
        #pragma unroll
        for (int v=0; v<8; v++){
          int r = v + 8*g;
          int i = qbase + r;
          int j = kbase + sub*16 + ln;
          int d = j - i;
          d = (d < -R) ? -R : ((d > R) ? R : d);
          atomicAdd(&wbin[wv][r*96 + d + R], sfr[sub][v]);
        }
      }
    }

    BF16Frag pf;
    pf.q[0] = *(const uint4*)&pst[wv][ln*32 + 16*g];
    pf.q[1] = *(const uint4*)&pst[wv][ln*32 + 16*g + 8];
    #pragma unroll
    for (int nc=0; nc<4; nc++){
      const uint4* vs = (const uint4*)(Vp + ((size_t)(((b*NH + h)*KT32_N + kb)*4 + nc))*512);
      BF16Frag vf; vf.q[0] = vs[lane*2]; vf.q[1] = vs[lane*2+1];
      ctx[nc] = wmma_bf16(pf, vf, ctx[nc]);
    }
  }

  __syncthreads();   // wbin final

  #pragma unroll
  for (int kc=0; kc<3; kc++){
    BF16Frag wf;
    #pragma unroll
    for (int e=0; e<16; e++){
      int kk = e + 16*g;
      wf.u[e] = f2bf(wbin[wv][ln*96 + kc*32 + kk]);
    }
    #pragma unroll
    for (int nt=0; nt<4; nt++){
      const uint4* es = (const uint4*)(EmbV + ((size_t)((seg*3 + kc)*4 + nt))*512);
      BF16Frag ev; ev.q[0] = es[lane*2]; ev.q[1] = es[lane*2+1];
      ctx[nt] = wmma_bf16(wf, ev, ctx[nt]);
    }
  }

  float rl[8];
  #pragma unroll
  for (int v=0; v<8; v++) rl[v] = 1.0f / lrow[v];
  #pragma unroll
  for (int nc=0; nc<4; nc++){
    #pragma unroll
    for (int v=0; v<8; v++){
      int r = v + 8*g;
      size_t row = (size_t)(b*STOT + qbase + r);
      int col = h*HDIM + nc*16 + ln;
      ctxbuf[row*DIMC + col] = f2bf(ctx[nc][v] * rl[v]);
    }
  }
}

// ===================================================================
// Kernel: output GEMM  out = ctx(bf16) @ Wout + bout  (f32 out)
// grid = (112, 8), block = 128
// ===================================================================
__global__ __launch_bounds__(128) void gemm_out_kernel(
    const unsigned short* __restrict__ ctxbf, const unsigned short* __restrict__ WoT,
    const float* __restrict__ bout, float* __restrict__ out)
{
  __shared__ __align__(16) unsigned short smem[SMEM_STAGE_USH];
  const int mb = blockIdx.x, nb = blockIdx.y;
  const unsigned short* Ablk = ctxbf + (size_t)mb*64*DIMC;
  const unsigned short* Bblk = WoT   + (size_t)nb*128*DIMC;

  v8f acc[2][4];
  #pragma unroll
  for (int mf=0; mf<2; mf++)
    #pragma unroll
    for (int nf=0; nf<4; nf++) acc[mf][nf] = vzero8();

  gemm_mainloop_bf(Ablk, Bblk, smem, acc);

  const int tid=threadIdx.x, wv=tid>>5, lane=tid&31, ln=lane&15, g=lane>>4;
  const int wm=(wv&1)*32, wn=(wv>>1)*64;
  #pragma unroll
  for (int mf=0; mf<2; mf++)
    #pragma unroll
    for (int nf=0; nf<4; nf++)
      #pragma unroll
      for (int v=0; v<8; v++){
        int gr = mb*64 + wm + mf*16 + v + 8*g;
        int gc = nb*128 + wn + nf*16 + ln;
        out[(size_t)gr*DIMC + gc] = acc[mf][nf][v] + bout[gc];
      }
}

// ===================================================================
extern "C" void kernel_launch(void* const* d_in, const int* in_sizes, int n_in,
                              void* d_out, int out_size, void* d_ws, size_t ws_size,
                              hipStream_t stream) {
  const float* stok = (const float*)d_in[0];
  const float* tok  = (const float*)d_in[1];
  const float* stm  = (const float*)d_in[2];
  const unsigned char* mk0 = (const unsigned char*)d_in[3];
  const unsigned char* mk1 = (const unsigned char*)d_in[4];
  const unsigned char* mk2 = (const unsigned char*)d_in[5];
  const float* Wqkv = (const float*)d_in[6];
  const float* bqkv = (const float*)d_in[7];
  const float* Wout = (const float*)d_in[8];
  const float* bout = (const float*)d_in[9];
  const float* emb_stok_k = (const float*)d_in[10];
  const float* emb_stok_v = (const float*)d_in[11];
  const float* emb_tok_k  = (const float*)d_in[12];
  const float* emb_tok_v  = (const float*)d_in[13];
  const float* emb_stm_k  = (const float*)d_in[14];
  const float* emb_stm_v  = (const float*)d_in[15];

  char* ws = (char*)d_ws;
  size_t off = 0;
  auto walloc = [&](size_t bytes) -> void* {
    void* p = ws + off;
    off += (bytes + 255) & ~(size_t)255;
    return p;
  };
  unsigned short* xbf  = (unsigned short*)walloc((size_t)NROWS*DIMC * 2);
  unsigned short* WqT  = (unsigned short*)walloc((size_t)3*DIMC*DIMC * 2);
  unsigned short* WoT  = (unsigned short*)walloc((size_t)DIMC*DIMC * 2);
  unsigned short* Qp   = (unsigned short*)walloc((size_t)BATCH*NH*QT_N*2   * 512 * 2);
  unsigned short* Kp   = (unsigned short*)walloc((size_t)BATCH*NH*QT_N*2   * 512 * 2);
  unsigned short* Vp   = (unsigned short*)walloc((size_t)BATCH*NH*KT32_N*4 * 512 * 2);
  unsigned short* EmbK = (unsigned short*)walloc((size_t)3*2*5 * 512 * 2);
  unsigned short* EmbV = (unsigned short*)walloc((size_t)3*3*4 * 512 * 2);
  unsigned short* ctxb = (unsigned short*)walloc((size_t)NROWS*DIMC * 2);

  pack_x_kernel<<<(NROWS*DIMC/8)/256, 256, 0, stream>>>(stok, tok, stm, xbf);
  pack_w_kernel<<<((3*DIMC*DIMC + DIMC*DIMC)/8)/256, 256, 0, stream>>>(Wqkv, Wout, WqT, WoT);
  pack_emb_kernel<<<132, 256, 0, stream>>>(
      emb_stok_k, emb_tok_k, emb_stm_k,
      emb_stok_v, emb_tok_v, emb_stm_v, EmbK, EmbV);

  gemm_qkv_kernel<<<dim3(112, 24), 128, 0, stream>>>(xbf, WqT, bqkv, Qp, Kp, Vp);

  attn_kernel<<<(BATCH*NH*QT_N)/4, 128, 0, stream>>>(
      Qp, Kp, Vp, EmbK, EmbV, mk0, mk1, mk2, ctxb);

  gemm_out_kernel<<<dim3(112, 8), 128, 0, stream>>>(ctxb, WoT, bout, (float*)d_out);
}
